// DTNN_54022098649352
// MI455X (gfx1250) — matrix operand order repaired and checked by reading, not compiled, not run on hardware
//
#include <hip/hip_runtime.h>
#include <cstdint>

// Problem constants (match reference)
#define B_  32
#define N_  64
#define NG_ 100
#define NB_ 128
#define NF_ 256

typedef __attribute__((ext_vector_type(16))) _Float16 v16h;
typedef __attribute__((ext_vector_type(8)))  float    v8f;
typedef __attribute__((ext_vector_type(2)))  _Float16 h2;

// ---------------------------------------------------------------------------
// CDNA5 async global->LDS copy (ASYNCcnt path). Guarded so compile never breaks.
// LDS DS-address = low 32 bits of the generic pointer (ISA 10.2 aperture rules).
// Builtin parameter type (from clang diagnostic): pointer to int vector_size(16).
// ---------------------------------------------------------------------------
typedef int vi4 __attribute__((vector_size(16)));
typedef __attribute__((address_space(1))) vi4 gvi4;
typedef __attribute__((address_space(3))) vi4 lvi4;

#if __has_builtin(__builtin_amdgcn_global_load_async_to_lds_b128)
#define HAS_ASYNC_LDS 1
__device__ __forceinline__ void async_copy_b128(const void* gsrc, void* ldst) {
  __builtin_amdgcn_global_load_async_to_lds_b128(
      (gvi4*)(uintptr_t)gsrc,
      (lvi4*)(unsigned)(uintptr_t)ldst,
      0, 0);
}
__device__ __forceinline__ void async_wait() {
#if __has_builtin(__builtin_amdgcn_s_wait_asynccnt)
  __builtin_amdgcn_s_wait_asynccnt(0);
#else
  asm volatile("s_wait_asynccnt 0x0" ::: "memory");
#endif
}
#else
#define HAS_ASYNC_LDS 0
__device__ __forceinline__ void async_copy_b128(const void* gsrc, void* ldst) {
  *(uint4*)ldst = *(const uint4*)gsrc; // fallback: regular staged copy
}
__device__ __forceinline__ void async_wait() {}
#endif

// ---------------------------------------------------------------------------
// WMMA helpers (CDNA5 wave32, 16x16x32 f16 -> f32 accumulate)
// ---------------------------------------------------------------------------
__device__ __forceinline__ v8f wmma_f16f32(v16h a, v16h b, v8f c) {
  // (neg_a, A, neg_b, B, c_mod, C, reuse_a, reuse_b)
  return __builtin_amdgcn_wmma_f32_16x16x32_f16(false, a, false, b, (short)0, c,
                                                false, false);
}

// A fragment: 16-bit A-matrix 16x32 (MxK).
// lane l: M = l&15 ; group g=l>>4 : VGPR v<4 -> K = 8g+2v+h ; v>=4 -> K = 16+8g+2(v-4)+h
__device__ __forceinline__ v16h load_a_frag(const _Float16* __restrict__ row,
                                            int k0, int lane) {
  const int g = (lane >> 4) << 3; // 0 or 8
  const h2* p0 = (const h2*)(row + k0 + g);
  const h2* p1 = (const h2*)(row + k0 + 16 + g);
  v16h a;
#pragma unroll
  for (int v = 0; v < 4; ++v) {
    h2 x = p0[v]; a[2 * v]     = x[0]; a[2 * v + 1]     = x[1];
    h2 y = p1[v]; a[8 + 2 * v] = y[0]; a[8 + 2 * v + 1] = y[1];
  }
  return a;
}

// B fragment: 16-bit B-matrix 32x16 (KxN).
// lane l: N = n0 + (l&15) ; g=l>>4 : s<8 -> K = k0+8g+s ; s>=8 -> K = k0+16+8g+(s-8)
__device__ __forceinline__ v16h load_b_frag(const _Float16* __restrict__ Bm,
                                            int ldb, int k0, int n0, int lane) {
  const int g = (lane >> 4) << 3;
  const _Float16* p = Bm + (size_t)k0 * ldb + n0 + (lane & 15);
  v16h b;
#pragma unroll
  for (int s = 0; s < 8; ++s) { b[s] = p[(size_t)(g + s) * ldb]; b[8 + s] = p[(size_t)(16 + g + s) * ldb]; }
  return b;
}

// ---------------------------------------------------------------------------
// k_prep: X = W_emb[Z]; convert Wc (zero-pad K 100->128), Wi, Wf to f16
// ---------------------------------------------------------------------------
__global__ void __launch_bounds__(256) k_prep(const int* __restrict__ Z,
                                              const float* __restrict__ W_emb,
                                              const float* __restrict__ Wc,
                                              const float* __restrict__ Wi,
                                              const float* __restrict__ Wf,
                                              float* __restrict__ X,
                                              _Float16* __restrict__ Wc_h,
                                              _Float16* __restrict__ Wi_h,
                                              _Float16* __restrict__ Wf_h) {
  const int tid = blockIdx.x * 256 + threadIdx.x;
  const int stride = gridDim.x * 256;
  for (int idx = tid; idx < B_ * N_ * NB_; idx += stride) {
    const int bn = idx >> 7, f = idx & 127;
    X[idx] = W_emb[Z[bn] * NB_ + f];
  }
  for (int idx = tid; idx < 128 * NF_; idx += stride) {   // Wc padded 128xNF
    const int g = idx >> 8;
    Wc_h[idx] = (g < NG_) ? (_Float16)Wc[g * NF_ + (idx & 255)] : (_Float16)0.f;
  }
  for (int idx = tid; idx < NB_ * NF_; idx += stride) Wi_h[idx] = (_Float16)Wi[idx];
  for (int idx = tid; idx < NF_ * NB_; idx += stride) Wf_h[idx] = (_Float16)Wf[idx];
}

// ---------------------------------------------------------------------------
// k_gemm1: fC[b,i,j,f] = C[b,i,j,:] @ Wc + bc   (stored f16)
// grid: x = b*64+i (2048), y = n-half (2).  K padded to 128 (4 wmma steps).
// Wc half tile staged via async global->LDS b128 copies.
// ---------------------------------------------------------------------------
__global__ void __launch_bounds__(256) k_gemm1(const float* __restrict__ C,
                                               const _Float16* __restrict__ Wc_h,
                                               const float* __restrict__ bc,
                                               _Float16* __restrict__ fC_h) {
  __shared__ _Float16 sWc[128 * 128]; // [K=128][Nhalf=128]  32 KB
  __shared__ _Float16 sC[64 * 128];   // [j=64][K=128]       16 KB
  const int tid = threadIdx.x;
  const int nh = blockIdx.y;
  const int lane = tid & 31, w = tid >> 5;

  // 128 rows x 128 halves; global row stride NF_=256 halves. 16 b128 chunks/row.
  for (int c = tid; c < 128 * 16; c += 256)
    async_copy_b128(Wc_h + (c >> 4) * NF_ + nh * 128 + (c & 15) * 8,
                    sWc + c * 8);
  const float* Crow = C + (size_t)blockIdx.x * N_ * NG_;
  for (int idx = tid; idx < 64 * 128; idx += 256) {
    const int j = idx >> 7, g = idx & 127;
    sC[idx] = (g < NG_) ? (_Float16)Crow[j * NG_ + g] : (_Float16)0.f;
  }
  async_wait();
  __syncthreads();

  const int nt = w;                               // wave -> fixed n-tile
  const int ncol = nh * 128 + nt * 16 + (lane & 15);
  const float bias = bc[ncol];
#pragma unroll
  for (int mt = 0; mt < 4; ++mt) {                // j-tiles
    v8f acc;
#pragma unroll
    for (int r = 0; r < 8; ++r) acc[r] = bias;
    const _Float16* arow = sC + (mt * 16 + (lane & 15)) * 128;
#pragma unroll
    for (int k = 0; k < 4; ++k) {
      v16h a = load_a_frag(arow, 32 * k, lane);
      v16h b = load_b_frag(sWc, 128, 32 * k, nt * 16, lane);
      acc = wmma_f16f32(a, b, acc);
    }
    const size_t rb = (size_t)blockIdx.x * N_ + mt * 16;
#pragma unroll
    for (int r = 0; r < 8; ++r) {
      const int M = ((lane >> 4) << 3) + r;
      fC_h[(rb + M) * NF_ + ncol] = (_Float16)acc[r];
    }
  }
}

// ---------------------------------------------------------------------------
// k_fx: fX[b,n,f] = X[b,n,:] @ Wi + bi  (stored f16)
// grid: x = row-tile over B*N (128), y = n-half (2)
// ---------------------------------------------------------------------------
__global__ void __launch_bounds__(256) k_fx(const float* __restrict__ X,
                                            const _Float16* __restrict__ Wi_h,
                                            const float* __restrict__ bi,
                                            _Float16* __restrict__ fX_h) {
  __shared__ _Float16 sWi[128 * 128]; // 32 KB
  __shared__ _Float16 sX[16 * 128];   //  4 KB
  const int tid = threadIdx.x;
  const int nh = blockIdx.y;
  const int lane = tid & 31, w = tid >> 5;

  for (int c = tid; c < 128 * 16; c += 256)
    async_copy_b128(Wi_h + (c >> 4) * NF_ + nh * 128 + (c & 15) * 8,
                    sWi + c * 8);
  for (int idx = tid; idx < 16 * 128; idx += 256)
    sX[idx] = (_Float16)X[((size_t)blockIdx.x * 16 + (idx >> 7)) * NB_ + (idx & 127)];
  async_wait();
  __syncthreads();

  const int nt = w;
  const int ncol = nh * 128 + nt * 16 + (lane & 15);
  const float bias = bi[ncol];
  v8f acc;
#pragma unroll
  for (int r = 0; r < 8; ++r) acc[r] = bias;
  const _Float16* arow = sX + (lane & 15) * 128;
#pragma unroll
  for (int k = 0; k < 4; ++k) {
    v16h a = load_a_frag(arow, 32 * k, lane);
    v16h b = load_b_frag(sWi, 128, 32 * k, nt * 16, lane);
    acc = wmma_f16f32(a, b, acc);
  }
#pragma unroll
  for (int r = 0; r < 8; ++r) {
    const int M = ((lane >> 4) << 3) + r;
    fX_h[((size_t)blockIdx.x * 16 + M) * NF_ + ncol] = (_Float16)acc[r];
  }
}

// ---------------------------------------------------------------------------
// k_inter: X[b,i,:] += sum_j mask[b,i,j] * tanh((fX[b,j,:]*fC[b,i,j,:]) @ Wf)
// grid: x = b*4 + itile (128), y = output-half (2). Wave w owns i = itile*16+2w+ii.
// fC stays L2-resident (192MB L2); Wf half + whole fX[b] staged in LDS (64 KB)
// via async global->LDS copies.
// ---------------------------------------------------------------------------
__global__ void __launch_bounds__(256) k_inter(const int* __restrict__ Z,
                                               const _Float16* __restrict__ fC_h,
                                               const _Float16* __restrict__ fX_h,
                                               const _Float16* __restrict__ Wf_h,
                                               float* __restrict__ X) {
  __shared__ _Float16 sWf[256 * 64]; // [K=256][Ohalf=64]  32 KB
  __shared__ _Float16 sFX[64 * 256]; // fX[b] tile         32 KB
  const int tid = threadIdx.x;
  const int b = blockIdx.x >> 2, itile = blockIdx.x & 3;
  const int nh = blockIdx.y;
  const int lane = tid & 31, w = tid >> 5;

  // Wf: 256 rows x 64 halves (global row stride NB_=128); 8 b128 chunks/row.
  for (int c = tid; c < 256 * 8; c += 256)
    async_copy_b128(Wf_h + (c >> 3) * NB_ + nh * 64 + (c & 7) * 8,
                    sWf + c * 8);
  // fX[b]: contiguous 64*256 halves = 2048 b128 chunks.
  const _Float16* fXb = fX_h + (size_t)b * N_ * NF_;
  for (int c = tid; c < 2048; c += 256)
    async_copy_b128(fXb + c * 8, sFX + c * 8);
  async_wait();
  __syncthreads();

  const int* Zb = Z + b * N_;
#pragma unroll
  for (int ii = 0; ii < 2; ++ii) {
    const int i = itile * 16 + 2 * w + ii;
    const int zi = __builtin_amdgcn_readfirstlane(Zb[i]); // wave-uniform branch
    float vacc[4] = {0.f, 0.f, 0.f, 0.f};
    if (zi > 0) {
      for (int jblk = 0; jblk < 4; ++jblk) {
        const int jr = jblk * 16 + (lane & 15); // A row (j) for this lane
        const _Float16* fCrow = fC_h + ((size_t)(b * N_ + i) * N_ + jr) * NF_;
        const _Float16* fXrow = sFX + jr * NF_;
        float m[8];
#pragma unroll
        for (int r = 0; r < 8; ++r) {
          const int j = jblk * 16 + ((lane >> 4) << 3) + r; // D row per lane
          m[r] = (Zb[j] > 0 && j != i) ? 1.f : 0.f;
        }
        v8f acc0 = {}, acc1 = {}, acc2 = {}, acc3 = {};
#pragma unroll
        for (int k = 0; k < 8; ++k) {
          v16h afc = load_a_frag(fCrow, 32 * k, lane);
          v16h afx = load_a_frag(fXrow, 32 * k, lane);
          v16h a = afc * afx; // fVj = fX_j * fC_ij, in f16
          acc0 = wmma_f16f32(a, load_b_frag(sWf, 64, 32 * k,  0, lane), acc0);
          acc1 = wmma_f16f32(a, load_b_frag(sWf, 64, 32 * k, 16, lane), acc1);
          acc2 = wmma_f16f32(a, load_b_frag(sWf, 64, 32 * k, 32, lane), acc2);
          acc3 = wmma_f16f32(a, load_b_frag(sWf, 64, 32 * k, 48, lane), acc3);
        }
        v8f* accs[4] = {&acc0, &acc1, &acc2, &acc3};
#pragma unroll
        for (int nt = 0; nt < 4; ++nt) {
          float s = 0.f;
#pragma unroll
          for (int r = 0; r < 8; ++r) s += tanhf((*accs[nt])[r]) * m[r];
          s += __shfl_xor(s, 16, 32); // combine j-halves (lanes l and l^16, same N)
          vacc[nt] += s;
        }
      }
    }
    if (lane < 16) {
#pragma unroll
      for (int nt = 0; nt < 4; ++nt)
        X[((size_t)(b * N_) + i) * NB_ + nh * 64 + nt * 16 + lane] += vacc[nt];
    }
  }
}

// ---------------------------------------------------------------------------
// k_readout: y[b] = sum_n (Z>0) * (tanh(X@W1+b1) @ W2 + b2). Deterministic.
// ---------------------------------------------------------------------------
__global__ void __launch_bounds__(256) k_readout(const int* __restrict__ Z,
                                                 const float* __restrict__ X,
                                                 const float* __restrict__ W1,
                                                 const float* __restrict__ b1,
                                                 const float* __restrict__ W2,
                                                 const float* __restrict__ b2,
                                                 float* __restrict__ out) {
  __shared__ float part[256];
  const int b = blockIdx.x, t = threadIdx.x;
  const int n = t >> 2, hq = t & 3; // thread -> (atom n, quarter of hidden units)
  const float* xr = X + ((size_t)b * N_ + n) * NB_;
  float p = 0.f;
  for (int h = hq * 16; h < hq * 16 + 16; ++h) {
    float d = b1[h];
#pragma unroll 4
    for (int f = 0; f < NB_; ++f) d += xr[f] * W1[f * (NB_ / 2) + h];
    p += tanhf(d) * W2[h];
  }
  part[t] = p;
  __syncthreads();
  if (t == 0) {
    const float bb2 = b2[0];
    float y = 0.f;
    for (int n2 = 0; n2 < N_; ++n2)
      if (Z[b * N_ + n2] > 0)
        y += part[4 * n2] + part[4 * n2 + 1] + part[4 * n2 + 2] + part[4 * n2 + 3] + bb2;
    out[b] = y;
  }
}

// ---------------------------------------------------------------------------
// Launch.  Workspace layout (bytes):
//   fC_h : 32*64*64*256*2 = 67,108,864   (fits in 192MB L2 -> 3 cheap re-reads)
//   fX_h : 2048*256*2     =  1,048,576
//   X    : 2048*128*4     =  1,048,576
//   Wc_h/Wi_h/Wf_h        =  3*65,536
// total ~69.4 MB required in d_ws.
// ---------------------------------------------------------------------------
extern "C" void kernel_launch(void* const* d_in, const int* in_sizes, int n_in,
                              void* d_out, int out_size, void* d_ws, size_t ws_size,
                              hipStream_t stream) {
  const int*   Z     = (const int*)d_in[0];
  const float* C     = (const float*)d_in[1];
  const float* W_emb = (const float*)d_in[2];
  const float* Wc    = (const float*)d_in[3];
  const float* bc    = (const float*)d_in[4];
  const float* Wi    = (const float*)d_in[5];
  const float* bi    = (const float*)d_in[6];
  const float* Wf    = (const float*)d_in[7];
  const float* W1    = (const float*)d_in[8];
  const float* b1    = (const float*)d_in[9];
  const float* W2    = (const float*)d_in[10];
  const float* b2    = (const float*)d_in[11];

  char* ws = (char*)d_ws;
  _Float16* fC_h = (_Float16*)ws;
  _Float16* fX_h = (_Float16*)(ws + 67108864);
  float*    X    = (float*)   (ws + 67108864 + 1048576);
  _Float16* Wc_h = (_Float16*)(ws + 67108864 + 1048576 + 1048576);
  _Float16* Wi_h = (_Float16*)(ws + 67108864 + 1048576 + 1048576 + 65536);
  _Float16* Wf_h = (_Float16*)(ws + 67108864 + 1048576 + 1048576 + 131072);

  k_prep<<<dim3(512), 256, 0, stream>>>(Z, W_emb, Wc, Wi, Wf, X, Wc_h, Wi_h, Wf_h);
  k_gemm1<<<dim3(B_ * N_, 2), 256, 0, stream>>>(C, Wc_h, bc, fC_h);
  for (int t = 0; t < 3; ++t) {
    k_fx<<<dim3((B_ * N_) / 16, 2), 256, 0, stream>>>(X, Wi_h, bi, fX_h);
    k_inter<<<dim3(B_ * (N_ / 16), 2), 256, 0, stream>>>(Z, fC_h, fX_h, Wf_h, X);
  }
  k_readout<<<dim3(B_), 256, 0, stream>>>(Z, X, W1, b1, W2, b2, (float*)d_out);
}
